// BiMamba2_76888504533061
// MI455X (gfx1250) — matrix operand order repaired and checked by reading, not compiled, not run on hardware
//
#include <hip/hip_runtime.h>
#include <hip/hip_bf16.h>
#include <math.h>
#include <stdint.h>

// ---------------- problem constants ----------------
#define B_SZ      4
#define L_SEQ     2048
#define DMODEL    1024
#define DSTATE    64
#define DCONV     4
#define HEADDIM   64
#define DINNER    2048
#define NHEADS    32
#define CONVDIM   2176            // DINNER + 2*DSTATE
#define DINPROJ   4256            // 2*DINNER + 2*DSTATE + NHEADS
#define ROWS      (B_SZ * L_SEQ)  // 8192

typedef unsigned short u16;
typedef unsigned int   u32;

typedef __attribute__((ext_vector_type(16))) __bf16 v16bf;
typedef __attribute__((ext_vector_type(8)))  float  v8f;

struct __align__(16) U128 { u32 x[4]; };
union FragBF  { v16bf v; U128 q[2]; };
union Pack128 { U128 u; u16 h[8]; };

__device__ __forceinline__ u16 f2bf(float f) {
  union { float f; u32 u; } c; c.f = f;
  u32 u = c.u;
  u32 r = u + 0x7FFFu + ((u >> 16) & 1u);   // round-to-nearest-even
  return (u16)(r >> 16);
}
__device__ __forceinline__ float silu_f(float x) { return x / (1.f + __expf(-x)); }

// ---- CDNA5 async global->LDS copy (ASYNCcnt-tracked, no VGPR data path) ----
// INST_OFFSET is added to BOTH the global and the LDS address (ISA 08_async_tensor
// §4.4), so a contiguous 64B row copy is 4 async b128s off one base pair.
// Generic-pointer low 32 bits == LDS byte offset (ISA §10.2 aperture rules).
__device__ __forceinline__ void async_copy_row64(void* lds_dst, const void* gsrc) {
  u32 dst = (u32)(uintptr_t)lds_dst;
  unsigned long long src = (unsigned long long)(uintptr_t)gsrc;
  asm volatile(
      "global_load_async_to_lds_b128 %0, %1, off\n\t"
      "global_load_async_to_lds_b128 %0, %1, off offset:16\n\t"
      "global_load_async_to_lds_b128 %0, %1, off offset:32\n\t"
      "global_load_async_to_lds_b128 %0, %1, off offset:48"
      :
      : "v"(dst), "v"(src)
      : "memory");
}
__device__ __forceinline__ void wait_async_all() {
  asm volatile("s_wait_asynccnt 0x0" ::: "memory");
}

// ---------------- f32 -> bf16 convert ----------------
__global__ void cvt_bf16_kernel(const float* __restrict__ in, u16* __restrict__ out, int n) {
  int i = blockIdx.x * blockDim.x + threadIdx.x;
  if (i < n) out[i] = f2bf(in[i]);
}

// ---------------- bf16 WMMA GEMM: C[M,N] = A[M,K] * B[K,N] (+bias) ----------------
// A row-major bf16 (lda), B row-major bf16 (ldb), C f32 (ldc).
// Block tile 256x128, 256 threads = 8 waves in 4(m) x 2(n); wave tile 64x64.
// LDS double-buffered; A staged via async global->LDS DMA; B staged transposed
// ([n][k]) with dword pair-packing.
__global__ __launch_bounds__(256) void gemm_bf16_wmma(
    const u16* __restrict__ A, int lda,
    const u16* __restrict__ Bw, int ldb,
    float* __restrict__ C, int ldc,
    int M, int N, int K, const float* __restrict__ bias)
{
  __shared__ __align__(16) u16 As[2][256][40];   // [m][k], +8 pad -> conflict-free b128
  __shared__ __align__(16) u16 Bs[2][128][40];   // [n][k] transposed

  const int tid  = threadIdx.x;
  const int lane = tid & 31;
  const int wid  = tid >> 5;
  const int wm   = (wid >> 1) * 64;   // wave M offset (0/64/128/192)
  const int wn   = (wid & 1) * 64;    // wave N offset (0/64)
  const int m0   = blockIdx.x * 256;
  const int n0   = blockIdx.y * 128;
  const int lmod = lane & 15;
  const int lhi  = lane >> 4;         // half-wave select
  const bool fullN = (n0 + 128) <= N; // all but the last ragged in_proj tile

  v8f acc[4][4];
  #pragma unroll
  for (int i = 0; i < 4; ++i)
    #pragma unroll
    for (int j = 0; j < 4; ++j)
      #pragma unroll
      for (int e = 0; e < 8; ++e) acc[i][j][e] = 0.f;

  // --- staging helpers -------------------------------------------------------
  auto stageA = [&](int sb, int kk) {
    // thread t stages one full A row: 32 bf16 = 64B, async DMA straight to LDS
    const u16* src = A + (size_t)(m0 + tid) * lda + kk;
    async_copy_row64(&As[sb][tid][0], src);
  };
  auto stageB = [&](int sb, int kk) {
    if (fullN) {
      // fast path: two adjacent K rows, 8 n's each; pack (k,k+1) pairs -> b32 stores
      const int k2 = (tid >> 4) << 1;       // 0,2,...,30
      const int n8 = (tid & 15) * 8;        // 0..120
      const u16* src = Bw + (size_t)(kk + k2) * ldb + (n0 + n8);
      Pack128 lo, hi;
      lo.u = *(const U128*)src;
      hi.u = *(const U128*)(src + ldb);
      #pragma unroll
      for (int i = 0; i < 8; ++i) {
        u32 pk = (u32)lo.h[i] | ((u32)hi.h[i] << 16);
        *(u32*)&Bs[sb][n8 + i][k2] = pk;
      }
      if (kk + 32 < K) __builtin_prefetch(src + (size_t)32 * ldb, 0, 1);  // global_prefetch_b8
    } else {
      // ragged tile: guarded element loop (1 of 34 tiles in in_proj only)
      const int k    = tid >> 3;
      const int nseg = (tid & 7) * 16;
      const u16* src = Bw + (size_t)(kk + k) * ldb + n0 + nseg;
      #pragma unroll
      for (int i = 0; i < 16; ++i) {
        u16 v = ((n0 + nseg + i) < N) ? src[i] : (u16)0;
        Bs[sb][nseg + i][k] = v;
      }
    }
  };

  // --- prologue + double-buffered main loop ---------------------------------
  stageA(0, 0);
  stageB(0, 0);
  int buf = 0;
  for (int k0 = 0; k0 < K; k0 += 32) {
    wait_async_all();                      // this wave's async LDS writes landed
    __syncthreads();                       // staging of `buf` complete, prev compute done
    const int kn = k0 + 32;
    if (kn < K) { stageA(buf ^ 1, kn); stageB(buf ^ 1, kn); }

    // Fragment loads per documented 16-bit WMMA VGPR layouts.
    // A (16x32): lanes 0-15 -> K[0..7],K[16..23]; lanes 16-31 -> K[8..15],K[24..31]
    // B (32x16): lanes 0-15 -> K[0..7],K[16..23]; lanes 16-31 -> K[8..15],K[24..31]
    FragBF fa[4], fb[4];
    #pragma unroll
    for (int mf = 0; mf < 4; ++mf) {
      const u16* p = &As[buf][wm + mf * 16 + lmod][lhi * 8];
      fa[mf].q[0] = *(const U128*)p;
      fa[mf].q[1] = *(const U128*)(p + 16);
    }
    #pragma unroll
    for (int nf = 0; nf < 4; ++nf) {
      const u16* p = &Bs[buf][wn + nf * 16 + lmod][lhi * 8];
      fb[nf].q[0] = *(const U128*)p;
      fb[nf].q[1] = *(const U128*)(p + 16);
    }
    #pragma unroll
    for (int mf = 0; mf < 4; ++mf)
      #pragma unroll
      for (int nf = 0; nf < 4; ++nf)
        acc[mf][nf] = __builtin_amdgcn_wmma_f32_16x16x32_bf16(
            false, fa[mf].v, false, fb[nf].v, (short)0, acc[mf][nf], false, false);
    buf ^= 1;
  }

  // --- store: C/D layout -> lane: N = lmod, VGPR v: M = v + 8*lhi -----------
  #pragma unroll
  for (int mf = 0; mf < 4; ++mf) {
    #pragma unroll
    for (int nf = 0; nf < 4; ++nf) {
      const int n = n0 + wn + nf * 16 + lmod;
      if (fullN || n < N) {
        const float bv = bias ? bias[n] : 0.f;
        #pragma unroll
        for (int v = 0; v < 8; ++v) {
          const int m = m0 + wm + mf * 16 + lhi * 8 + v;
          C[(size_t)m * ldc + n] = acc[mf][nf][v] + bv;
        }
      }
    }
  }
}

// ---------------- depthwise causal conv (width 4) + bias + SiLU ----------------
__global__ void conv_silu_kernel(const float* __restrict__ zx,
                                 const float* __restrict__ conv_w,
                                 const float* __restrict__ conv_b,
                                 float* __restrict__ xBC, int rev)
{
  int idx = blockIdx.x * blockDim.x + threadIdx.x;
  const int total = B_SZ * L_SEQ * CONVDIM;
  if (idx >= total) return;
  int c = idx % CONVDIM;
  int t = (idx / CONVDIM) % L_SEQ;
  int b = idx / (CONVDIM * L_SEQ);
  float accv = conv_b[c];
  #pragma unroll
  for (int k = 0; k < DCONV; ++k) {
    int s = t - (DCONV - 1) + k;
    if (s >= 0) {
      int so = rev ? (L_SEQ - 1 - s) : s;
      accv += zx[(size_t)(b * L_SEQ + so) * DINPROJ + DINNER + c] * conv_w[c * DCONV + k];
    }
  }
  xBC[(size_t)idx] = silu_f(accv);
}

// ---------------- sequential SSM scan: one workgroup per (batch, head) ----------------
__global__ __launch_bounds__(256) void scan_kernel(
    const float* __restrict__ zx, const float* __restrict__ xBC,
    const float* __restrict__ dt_bias, const float* __restrict__ A_log,
    const float* __restrict__ Dp, float* __restrict__ y, int rev)
{
  const int b = blockIdx.x / NHEADS;
  const int h = blockIdx.x % NHEADS;
  const int tid = threadIdx.x;
  const int q = tid >> 6;     // n-quarter 0..3
  const int p = tid & 63;     // head-dim index
  const float Ah  = -__expf(A_log[h]);
  const float dtb = dt_bias[h];
  const float Dh  = Dp[h];

  __shared__ float sB[64], sC[64], sX[64], sScal[2];
  __shared__ float sRed[4][64];

  float s[16];
  #pragma unroll
  for (int i = 0; i < 16; ++i) s[i] = 0.f;

  for (int t = 0; t < L_SEQ; ++t) {
    const int to = rev ? (L_SEQ - 1 - t) : t;
    const size_t rx = (size_t)(b * L_SEQ + t) * CONVDIM;
    if (tid < 64)       sB[tid]       = xBC[rx + DINNER + tid];
    else if (tid < 128) sC[tid - 64]  = xBC[rx + DINNER + DSTATE + (tid - 64)];
    else if (tid < 192) sX[tid - 128] = xBC[rx + h * HEADDIM + (tid - 128)];
    else if (tid == 192) {
      float raw = zx[(size_t)(b * L_SEQ + to) * DINPROJ + (DINPROJ - NHEADS) + h] + dtb;
      float dt  = (raw > 20.f) ? raw : log1pf(__expf(raw));  // softplus
      sScal[0] = dt;
      sScal[1] = __expf(dt * Ah);                            // dA
    }
    __syncthreads();
    const float dt   = sScal[0];
    const float dA   = sScal[1];
    const float xp   = sX[p];
    const float coef = dt * xp;
    float accv = 0.f;
    #pragma unroll
    for (int i = 0; i < 16; ++i) {
      const float bn = sB[q * 16 + i];
      const float cn = sC[q * 16 + i];
      s[i]  = fmaf(s[i], dA, coef * bn);      // S = S*dA + (dt*x) B
      accv  = fmaf(s[i], cn, accv);           // y += S*C
    }
    sRed[q][p] = accv;
    __syncthreads();
    if (tid < 64) {   // p == tid here, xp already holds sX[tid]
      float yv = sRed[0][tid] + sRed[1][tid] + sRed[2][tid] + sRed[3][tid] + Dh * xp;
      y[(size_t)(b * L_SEQ + to) * DINNER + h * HEADDIM + tid] = yv;
    }
  }
}

// ---------------- gate with silu(z), RMS-norm, emit bf16 ----------------
__global__ __launch_bounds__(256) void gate_rms_kernel(
    const float* __restrict__ y_f, const float* __restrict__ y_b,
    const float* __restrict__ zx, const float* __restrict__ norm_w,
    u16* __restrict__ g_f, u16* __restrict__ g_b)
{
  const int row = blockIdx.x;
  const int dir = blockIdx.y;
  const float* yr = (dir ? y_b : y_f) + (size_t)row * DINNER;
  u16*         gr = (dir ? g_b : g_f) + (size_t)row * DINNER;
  const float* zr = zx + (size_t)row * DINPROJ;
  const int tid = threadIdx.x;

  float v[8]; float ss = 0.f;
  #pragma unroll
  for (int j = 0; j < 8; ++j) {
    int c = j * 256 + tid;
    float g = v[j] = yr[c] * silu_f(zr[c]);
    ss += g * g;
  }
  __shared__ float red[256];
  red[tid] = ss; __syncthreads();
  for (int st = 128; st > 0; st >>= 1) {
    if (tid < st) red[tid] += red[tid + st];
    __syncthreads();
  }
  const float scale = rsqrtf(red[0] * (1.f / DINNER) + 1e-5f);
  #pragma unroll
  for (int j = 0; j < 8; ++j) {
    int c = j * 256 + tid;
    gr[c] = f2bf(v[j] * scale * norm_w[c]);
  }
}

// ---------------- host side ----------------
extern "C" void kernel_launch(void* const* d_in, const int* in_sizes, int n_in,
                              void* d_out, int out_size, void* d_ws, size_t ws_size,
                              hipStream_t stream) {
  (void)in_sizes; (void)n_in; (void)out_size; (void)ws_size;
  const float* x          = (const float*)d_in[0];
  const float* in_proj_w  = (const float*)d_in[1];
  const float* conv_w     = (const float*)d_in[2];
  const float* conv_b     = (const float*)d_in[3];
  const float* dt_bias    = (const float*)d_in[4];
  const float* A_log      = (const float*)d_in[5];
  const float* Dp         = (const float*)d_in[6];
  const float* norm_w     = (const float*)d_in[7];
  const float* out_proj_w = (const float*)d_in[8];
  const float* proj_w     = (const float*)d_in[9];
  const float* proj_b     = (const float*)d_in[10];
  float* out = (float*)d_out;

  char* base = (char*)d_ws;
  size_t off = 0;
  auto alloc = [&](size_t bytes) -> char* {
    char* p = base + off;
    off = (off + bytes + 255) & ~(size_t)255;
    return p;
  };

  float* zx    = (float*)alloc((size_t)ROWS * DINPROJ * 4);   // shared by fwd+bwd
  float* xBC_f = (float*)alloc((size_t)ROWS * CONVDIM * 4);
  float* xBC_b = (float*)alloc((size_t)ROWS * CONVDIM * 4);
  float* y_f   = (float*)alloc((size_t)ROWS * DINNER * 4);
  float* y_b   = (float*)alloc((size_t)ROWS * DINNER * 4);
  u16*   xb    = (u16*)alloc((size_t)ROWS * DMODEL * 2);
  u16*   wib   = (u16*)alloc((size_t)DMODEL * DINPROJ * 2);
  u16*   wob   = (u16*)alloc((size_t)DINNER * DMODEL * 2);
  u16*   wpb   = (u16*)alloc((size_t)(2 * DMODEL) * DMODEL * 2);
  u16*   g_f   = (u16*)alloc((size_t)ROWS * DINNER * 2);
  u16*   g_b   = (u16*)alloc((size_t)ROWS * DINNER * 2);
  u16*   catb  = (u16*)alloc((size_t)ROWS * (2 * DMODEL) * 2);
  float* catf  = y_f;   // alias: y_f is dead after gate_rms, reuse as f32 concat buffer

  // 1) bf16 conversions
  cvt_bf16_kernel<<<(ROWS * DMODEL + 255) / 256, 256, 0, stream>>>(x, xb, ROWS * DMODEL);
  cvt_bf16_kernel<<<(DMODEL * DINPROJ + 255) / 256, 256, 0, stream>>>(in_proj_w, wib, DMODEL * DINPROJ);
  cvt_bf16_kernel<<<(DINNER * DMODEL + 255) / 256, 256, 0, stream>>>(out_proj_w, wob, DINNER * DMODEL);
  cvt_bf16_kernel<<<(2 * DMODEL * DMODEL + 255) / 256, 256, 0, stream>>>(proj_w, wpb, 2 * DMODEL * DMODEL);

  // 2) in_proj GEMM (computed once; bwd direction is a row-reversal of this)
  dim3 g1(ROWS / 256, (DINPROJ + 127) / 128);
  gemm_bf16_wmma<<<g1, 256, 0, stream>>>(xb, DMODEL, wib, DINPROJ, zx, DINPROJ,
                                         ROWS, DINPROJ, DMODEL, nullptr);

  // 3) conv + SiLU, both directions
  const int convTotal = B_SZ * L_SEQ * CONVDIM;
  conv_silu_kernel<<<(convTotal + 255) / 256, 256, 0, stream>>>(zx, conv_w, conv_b, xBC_f, 0);
  conv_silu_kernel<<<(convTotal + 255) / 256, 256, 0, stream>>>(zx, conv_w, conv_b, xBC_b, 1);

  // 4) SSM scans (one WG per (b,h); 128 WGs per direction)
  scan_kernel<<<B_SZ * NHEADS, 256, 0, stream>>>(zx, xBC_f, dt_bias, A_log, Dp, y_f, 0);
  scan_kernel<<<B_SZ * NHEADS, 256, 0, stream>>>(zx, xBC_b, dt_bias, A_log, Dp, y_b, 1);

  // 5) gate + RMS norm -> bf16 (both directions in one launch)
  dim3 gg(ROWS, 2);
  gate_rms_kernel<<<gg, 256, 0, stream>>>(y_f, y_b, zx, norm_w, g_f, g_b);

  // 6) out_proj GEMMs into concatenated f32 buffer (fwd -> cols 0..1023, bwd -> 1024..2047)
  dim3 g2(ROWS / 256, DMODEL / 128);
  gemm_bf16_wmma<<<g2, 256, 0, stream>>>(g_f, DINNER, wob, DMODEL, catf, 2 * DMODEL,
                                         ROWS, DMODEL, DINNER, nullptr);
  gemm_bf16_wmma<<<g2, 256, 0, stream>>>(g_b, DINNER, wob, DMODEL, catf + DMODEL, 2 * DMODEL,
                                         ROWS, DMODEL, DINNER, nullptr);

  // 7) final projection: out = cat @ proj_w + proj_b
  cvt_bf16_kernel<<<(ROWS * 2 * DMODEL + 255) / 256, 256, 0, stream>>>(catf, catb, ROWS * 2 * DMODEL);
  gemm_bf16_wmma<<<g2, 256, 0, stream>>>(catb, 2 * DMODEL, wpb, DMODEL, out, DMODEL,
                                         ROWS, DMODEL, 2 * DMODEL, proj_b);
}
